// RGCN_6468220747933
// MI455X (gfx1250) — matrix operand order repaired and checked by reading, not compiled, not run on hardware
//
#include <hip/hip_runtime.h>

// ---------------------------------------------------------------------------
// RGCN on MI455X (gfx1250, wave32). WMMA bf16 for all dense math, f32 atomics
// for the (memory-bound) edge aggregation, global_prefetch for gather latency.
//
// Problem constants (match reference):
#define NNODES   50000
#define NRELS    5
#define NTYPES   3
#define DIN      256
#define DHID     256
#define DOUTF    128
#define EPW      4        // edges per wave in the scatter kernel
// ---------------------------------------------------------------------------

typedef __attribute__((ext_vector_type(16))) __bf16         v16bf;
typedef __attribute__((ext_vector_type(16))) unsigned short v16u;
typedef __attribute__((ext_vector_type(8)))  unsigned       v8u;
typedef __attribute__((ext_vector_type(8)))  float          v8f;

static __device__ __forceinline__ unsigned short bf16r(float f) {
    unsigned u = __builtin_bit_cast(unsigned, f);
    u += 0x7FFFu + ((u >> 16) & 1u);          // round-to-nearest-even
    return (unsigned short)(u >> 16);
}

#if __has_builtin(__builtin_amdgcn_cvt_pk_bf16_f32)
typedef __attribute__((ext_vector_type(2))) __bf16 v2bf;
static __device__ __forceinline__ unsigned pk_bf16(float a, float b) {
    v2bf p = __builtin_amdgcn_cvt_pk_bf16_f32(a, b);   // v_cvt_pk_bf16_f32
    return __builtin_bit_cast(unsigned, p);
}
#else
static __device__ __forceinline__ unsigned pk_bf16(float a, float b) {
    return (unsigned)bf16r(a) | ((unsigned)bf16r(b) << 16);
}
#endif

static __device__ __forceinline__ v8f wmma_bf16(v16bf a, v16bf b, v8f c) {
    // (neg_a, A, neg_b, B, c_mod, C, reuse_a, reuse_b)
    return __builtin_amdgcn_wmma_f32_16x16x32_bf16(false, a, false, b,
                                                   (short)0, c, false, false);
}

// A-fragment (16x32 bf16, MxK): lane m=lane&15, hf=lane>>4.
// elements e<8 : K = k0 + hf*8 + e        (contiguous run of 8 floats)
// elements e>=8: K = k0 + 16 + hf*8 + e-8 (contiguous run of 8 floats)
// p points at rowBase + k0 + hf*8 (32B aligned); scale applied before cvt.
static __device__ __forceinline__ v16bf load_a_f32(const float* __restrict__ p,
                                                   float s) {
    const float4* q = (const float4*)p;
    float4 f0 = q[0], f1 = q[1];   // K offsets +0..7
    float4 f2 = q[4], f3 = q[5];   // K offsets +16..23
    v8u u;
    u[0] = pk_bf16(f0.x * s, f0.y * s);
    u[1] = pk_bf16(f0.z * s, f0.w * s);
    u[2] = pk_bf16(f1.x * s, f1.y * s);
    u[3] = pk_bf16(f1.z * s, f1.w * s);
    u[4] = pk_bf16(f2.x * s, f2.y * s);
    u[5] = pk_bf16(f2.z * s, f2.w * s);
    u[6] = pk_bf16(f3.x * s, f3.y * s);
    u[7] = pk_bf16(f3.z * s, f3.w * s);
    return __builtin_bit_cast(v16bf, u);
}

// ---------------------------------------------------------------------------
// 0) Group input: h[n,:] = x_list[node_type[n]][local_idx[n],:]  (float4 copy)
// ---------------------------------------------------------------------------
__global__ void __launch_bounds__(256) group_input_kernel(
        const float* __restrict__ x0, const float* __restrict__ x1,
        const float* __restrict__ x2, const int* __restrict__ ntype,
        const int* __restrict__ lidx, float* __restrict__ h) {
    int tid = blockIdx.x * blockDim.x + threadIdx.x;
    if (tid >= NNODES * (DIN / 4)) return;
    int n = tid >> 6, c = tid & 63;
    int t = ntype[n];
    const float* src = (t == 0) ? x0 : (t == 1) ? x1 : x2;
    const float4* s4 = (const float4*)src;
    ((float4*)h)[(size_t)n * 64 + c] = s4[(size_t)lidx[n] * 64 + c];
}

// ---------------------------------------------------------------------------
// 1) Zero sums+cnt (contiguous region), float4 granularity
// ---------------------------------------------------------------------------
__global__ void __launch_bounds__(256) zero_f4_kernel(float4* __restrict__ p, int n4) {
    int i = blockIdx.x * blockDim.x + threadIdx.x;
    if (i < n4) p[i] = make_float4(0.f, 0.f, 0.f, 0.f);
}

// ---------------------------------------------------------------------------
// 2) Edge scatter: one wave handles EPW edges; 8 coalesced f32 atomics per
//    lane per edge; next edge's source row prefetched (global_prefetch_b8)
//    while the current edge's atomics drain.
// ---------------------------------------------------------------------------
__global__ void __launch_bounds__(256) edge_scatter_kernel(
        const float* __restrict__ h, const int* __restrict__ eidx,
        const int* __restrict__ etype, float* __restrict__ sums,
        float* __restrict__ cnt, int E) {
    int wid  = (int)((blockIdx.x * blockDim.x + threadIdx.x) >> 5);
    int lane = threadIdx.x & 31;
    int e0   = wid * EPW;
    if (e0 >= E) return;                   // wave-uniform exit
    int eEnd = (e0 + EPW < E) ? e0 + EPW : E;

    // prefetch first source row (lane strides of 32B cover the 1KB row)
    __builtin_prefetch(&h[(size_t)eidx[e0] * DIN + lane * 8], 0, 0);

    for (int e = e0; e < eEnd; ++e) {
        if (e + 1 < eEnd)
            __builtin_prefetch(&h[(size_t)eidx[e + 1] * DIN + lane * 8], 0, 0);
        int src = eidx[e];
        int dst = eidx[E + e];
        int r   = etype[e];
        const float* hrow = h + (size_t)src * DIN;
        float* srow = sums + ((size_t)dst * NRELS + r) * DIN;
#pragma unroll
        for (int i = 0; i < 8; ++i)
            atomicAdd(&srow[lane + i * 32], hrow[lane + i * 32]);
        if (lane == 0) atomicAdd(&cnt[dst * NRELS + r], 1.0f);
    }
}

// ---------------------------------------------------------------------------
// 3) f32 -> bf16 weight conversion (weights are tiny, done once per call)
// ---------------------------------------------------------------------------
__global__ void __launch_bounds__(256) cvt_bf16_kernel(
        const float* __restrict__ in, unsigned short* __restrict__ out, int n) {
    int i = blockIdx.x * blockDim.x + threadIdx.x;
    if (i < n) out[i] = bf16r(in[i]);
}

// ---------------------------------------------------------------------------
// 4) Fused RGCN layer GEMM (WMMA bf16):
//    out[n,o] = sum_r (sums[n,r,:]/max(cnt,1)) . Wrel[r][o,:]
//             + (x[n,:] . Wroot[type(n)][o,:]) + bias[type(n)][o]  (+ ReLU)
//    One wave computes a 16 x (16*NSUB) output slab: NSUB WMMA col subtiles
//    share each (expensive, f32-sourced) A fragment. Root transform merged
//    into the same accumulators by masking A rows per node type (EXEC stays
//    all-1s, as WMMA requires).
// ---------------------------------------------------------------------------
template <int NSUB>
__global__ void __launch_bounds__(256) rgcn_gemm_kernel(
        const float* __restrict__ sums, const float* __restrict__ cnt,
        const float* __restrict__ x,
        const unsigned short* __restrict__ Wrel,
        const unsigned short* __restrict__ Wroot,
        const float* __restrict__ bias, const int* __restrict__ ntype,
        float* __restrict__ out, int Dout, int ncg, int relu) {
    const int NT16 = NNODES / 16;
    int wid = (int)((blockIdx.x * blockDim.x + threadIdx.x) >> 5);
    int cg = wid % ncg;
    int rt = wid / ncg;
    if (rt >= NT16) return;                 // wave-uniform exit
    int lane = threadIdx.x & 31;
    int m    = lane & 15;
    int hf   = lane >> 4;
    int row0 = rt * 16;
    int col0 = cg * (16 * NSUB);
    int node = row0 + m;

    v8f acc[NSUB];
#pragma unroll
    for (int c4 = 0; c4 < NSUB; ++c4)
#pragma unroll
        for (int i = 0; i < 8; ++i) acc[c4][i] = 0.0f;

    // ---- relation messages: A = sums/max(cnt,1) (scaled during cvt) ----
#pragma unroll 1
    for (int r = 0; r < NRELS; ++r) {
        float inv = 1.0f / fmaxf(cnt[node * NRELS + r], 1.0f);
        const float* arow = sums + ((size_t)node * NRELS + r) * DIN;
        const unsigned short* wr = Wrel + (size_t)r * Dout * DIN;
#pragma unroll 1
        for (int k0 = 0; k0 < DIN; k0 += 32) {
            v16bf a = load_a_f32(arow + k0 + hf * 8, inv);
#pragma unroll
            for (int c4 = 0; c4 < NSUB; ++c4) {
                const unsigned short* bp =
                    wr + (size_t)(col0 + c4 * 16 + m) * DIN + k0 + hf * 8;
                typedef __attribute__((ext_vector_type(8))) __bf16 v8bf_;
                v8bf_ b0 = *(const v8bf_*)bp, b1 = *(const v8bf_*)(bp + 16);
                v16bf b = __builtin_shufflevector(b0, b1, 0,1,2,3,4,5,6,7,8,9,10,11,12,13,14,15);       // K hf*8+0..7 and 16+hf*8+0..7, 2x16B
                acc[c4] = wmma_bf16(a, b, acc[c4]);
            }
        }
    }

    // ---- root transform: mask A rows by node type, shared accumulator ----
    int nt = ntype[node];
    const float* xrow = x + (size_t)node * DIN;
#pragma unroll 1
    for (int t = 0; t < NTYPES; ++t) {
        float sc = (nt == t) ? 1.0f : 0.0f;
        const unsigned short* wt = Wroot + (size_t)t * Dout * DIN;
#pragma unroll 1
        for (int k0 = 0; k0 < DIN; k0 += 32) {
            v16bf a = load_a_f32(xrow + k0 + hf * 8, sc);
#pragma unroll
            for (int c4 = 0; c4 < NSUB; ++c4) {
                const unsigned short* bp =
                    wt + (size_t)(col0 + c4 * 16 + m) * DIN + k0 + hf * 8;
                typedef __attribute__((ext_vector_type(8))) __bf16 v8bf_;
                v8bf_ b0 = *(const v8bf_*)bp, b1 = *(const v8bf_*)(bp + 16);
                v16bf b = __builtin_shufflevector(b0, b1, 0,1,2,3,4,5,6,7,8,9,10,11,12,13,14,15);
                acc[c4] = wmma_bf16(a, b, acc[c4]);
            }
        }
    }

    // ---- epilogue: C layout -> (row = row0 + hf*8 + i, col = col0+c4*16+m)
#pragma unroll
    for (int c4 = 0; c4 < NSUB; ++c4) {
        int col = col0 + c4 * 16 + m;
#pragma unroll
        for (int i = 0; i < 8; ++i) {
            int row = row0 + hf * 8 + i;
            int t2  = ntype[row];
            float v = acc[c4][i] + bias[t2 * Dout + col];
            if (relu) v = fmaxf(v, 0.0f);
            out[(size_t)row * Dout + col] = v;
        }
    }
}

// ---------------------------------------------------------------------------
// 5) log_softmax over [N,128]: one wave per row, shfl_xor reductions
// ---------------------------------------------------------------------------
__global__ void __launch_bounds__(256) log_softmax_kernel(
        const float* __restrict__ in, float* __restrict__ out) {
    int wid  = (int)((blockIdx.x * blockDim.x + threadIdx.x) >> 5);
    int lane = threadIdx.x & 31;
    if (wid >= NNODES) return;
    const float* row = in + (size_t)wid * DOUTF;
    float v0 = row[lane], v1 = row[lane + 32];
    float v2 = row[lane + 64], v3 = row[lane + 96];
    float mx = fmaxf(fmaxf(v0, v1), fmaxf(v2, v3));
#pragma unroll
    for (int o = 16; o >= 1; o >>= 1) mx = fmaxf(mx, __shfl_xor(mx, o, 32));
    float s = __expf(v0 - mx) + __expf(v1 - mx) + __expf(v2 - mx) + __expf(v3 - mx);
#pragma unroll
    for (int o = 16; o >= 1; o >>= 1) s += __shfl_xor(s, o, 32);
    float L = mx + __logf(s);
    float* orow = out + (size_t)wid * DOUTF;
    orow[lane]      = v0 - L;
    orow[lane + 32] = v1 - L;
    orow[lane + 64] = v2 - L;
    orow[lane + 96] = v3 - L;
}

// ---------------------------------------------------------------------------
extern "C" void kernel_launch(void* const* d_in, const int* in_sizes, int n_in,
                              void* d_out, int out_size, void* d_ws, size_t ws_size,
                              hipStream_t stream) {
    // dict order: x0 x1 emb2 edge_index edge_type node_type local_node_idx
    //             rel_W0 root_W0 root_b0 rel_W1 root_W1 root_b1 rel_W2 root_W2 root_b2
    const float* x0   = (const float*)d_in[0];
    const float* x1   = (const float*)d_in[1];
    const float* emb2 = (const float*)d_in[2];
    const int* eidx   = (const int*)d_in[3];
    const int* etype  = (const int*)d_in[4];
    const int* ntype  = (const int*)d_in[5];
    const int* lidx   = (const int*)d_in[6];
    const float* relW[3]  = {(const float*)d_in[7],  (const float*)d_in[10], (const float*)d_in[13]};
    const float* rootW[3] = {(const float*)d_in[8],  (const float*)d_in[11], (const float*)d_in[14]};
    const float* rootB[3] = {(const float*)d_in[9],  (const float*)d_in[12], (const float*)d_in[15]};
    const int E = in_sizes[3] / 2;

    // ---- workspace layout (bump allocator, 256B aligned) ----
    char* base = (char*)d_ws;
    size_t off = 0;
    auto alloc = [&](size_t bytes) -> void* {
        void* p = base + off;
        off = (off + bytes + 255) & ~(size_t)255;
        return p;
    };
    float* hA = (float*)alloc((size_t)NNODES * DHID * sizeof(float));   // 51.2 MB
    float* hB = (float*)alloc((size_t)NNODES * DHID * sizeof(float));   // 51.2 MB
    const size_t sumsElems = (size_t)NNODES * NRELS * DIN;              // 64,000,000
    const size_t cntElems  = (size_t)NNODES * NRELS;                    // 250,000
    float* sums = (float*)alloc((sumsElems + cntElems) * sizeof(float)); // 257 MB
    float* cnt  = sums + sumsElems;                                      // contiguous
    const int relElems[3]  = {NRELS * DHID * DIN, NRELS * DHID * DIN, NRELS * DOUTF * DIN};
    const int rootElems[3] = {NTYPES * DHID * DIN, NTYPES * DHID * DIN, NTYPES * DOUTF * DIN};
    unsigned short* wrel[3];
    unsigned short* wroot[3];
    for (int l = 0; l < 3; ++l) {
        wrel[l]  = (unsigned short*)alloc((size_t)relElems[l] * 2);
        wroot[l] = (unsigned short*)alloc((size_t)rootElems[l] * 2);
    }

    // ---- convert weights to bf16 ----
    for (int l = 0; l < 3; ++l) {
        cvt_bf16_kernel<<<(relElems[l] + 255) / 256, 256, 0, stream>>>(relW[l], wrel[l], relElems[l]);
        cvt_bf16_kernel<<<(rootElems[l] + 255) / 256, 256, 0, stream>>>(rootW[l], wroot[l], rootElems[l]);
    }

    // ---- group input ----
    group_input_kernel<<<(NNODES * 64 + 255) / 256, 256, 0, stream>>>(
        x0, x1, emb2, ntype, lidx, hA);

    const int zeroN4 = (int)((sumsElems + cntElems) / 4);
    const int Douts[3] = {DHID, DHID, DOUTF};
    const float* xin[3]  = {hA, hB, hA};
    float*       xout[3] = {hB, hA, hB};

    const int scatterWaves  = (E + EPW - 1) / EPW;
    const int scatterBlocks = (scatterWaves + 7) / 8;

    for (int l = 0; l < 3; ++l) {
        zero_f4_kernel<<<(zeroN4 + 255) / 256, 256, 0, stream>>>((float4*)sums, zeroN4);
        edge_scatter_kernel<<<scatterBlocks, 256, 0, stream>>>(
            xin[l], eidx, etype, sums, cnt, E);
        const int ncg   = Douts[l] / (16 * 8);      // NSUB = 8 -> 128-col slabs
        const int waves = (NNODES / 16) * ncg;
        rgcn_gemm_kernel<8><<<(waves + 7) / 8, 256, 0, stream>>>(
            sums, cnt, xin[l], wrel[l], wroot[l], rootB[l], ntype,
            xout[l], Douts[l], ncg, (l != 2) ? 1 : 0);
    }

    // ---- log_softmax: hB holds [N,128] after layer 2 ----
    log_softmax_kernel<<<(NNODES + 7) / 8, 256, 0, stream>>>(hB, (float*)d_out);
}
